// ALIFSNN_28733331210303
// MI455X (gfx1250) — matrix operand order repaired and checked by reading, not compiled, not run on hardware
//
#include <hip/hip_runtime.h>

typedef __attribute__((ext_vector_type(16))) _Float16 v16h;
typedef __attribute__((ext_vector_type(8)))  _Float16 v8h;
typedef __attribute__((ext_vector_type(8)))  float    v8f;
typedef __attribute__((ext_vector_type(4)))  float    v4f;
typedef __attribute__((ext_vector_type(4)))  int      v4i;

#define T_STEPS 1024
#define B_DIM   256
#define I_DIM   256
#define H_DIM   512
#define O_DIM   128
#define IH_DIM  (I_DIM + H_DIM)   // 768

#define NBLK 64
#define TPB  256                  // 8 waves per block (wave32)
#define WPB  (TPB / 32)

// ---------------- workspace layout (bytes, all offsets 256-aligned) -------------
#define OFF_BAR  ((size_t)0)
#define OFF_U    ((size_t)256)
#define OFF_A    (OFF_U  + (size_t)B_DIM * H_DIM * 4)
#define OFF_OU   (OFF_A  + (size_t)B_DIM * H_DIM * 4)
#define OFF_Z0   (OFF_OU + (size_t)B_DIM * O_DIM * 4)
#define OFF_Z1   (OFF_Z0 + (size_t)B_DIM * H_DIM * 2)
#define OFF_WH   (OFF_Z1 + (size_t)B_DIM * H_DIM * 2)
#define OFF_WO   (OFF_WH + (size_t)H_DIM * IH_DIM * 2)
#define WS_BYTES (OFF_WO + (size_t)O_DIM * H_DIM * 2)

static_assert(NBLK * WPB == (B_DIM / 16) * (H_DIM / 16), "phase-A: one tile per wave");

// ---------------- gfx1250 async global->LDS copy (ASYNCcnt-tracked) -------------
#if __has_builtin(__builtin_amdgcn_global_load_async_to_lds_b128)
#define ASYNC_CP 1
__device__ __forceinline__ void async_cp_b128(const float* g, float* l) {
    __builtin_amdgcn_global_load_async_to_lds_b128(
        (__attribute__((address_space(1))) v4i*)(void*)g,
        (__attribute__((address_space(3))) v4i*)(void*)l, 0, 0);
}
#else
#define ASYNC_CP 0
__device__ __forceinline__ void async_cp_b128(const float* g, float* l) {
    *(v4f*)l = *(const v4f*)g;    // synchronous fallback
}
#endif

__device__ __forceinline__ void wait_async(int n) {
#if ASYNC_CP
#if __has_builtin(__builtin_amdgcn_s_wait_asynccnt)
    if (n == 0) __builtin_amdgcn_s_wait_asynccnt(0);
    else        __builtin_amdgcn_s_wait_asynccnt(4);
#else
    if (n == 0) asm volatile("s_wait_asynccnt 0" ::: "memory");
    else        asm volatile("s_wait_asynccnt 4" ::: "memory");
#endif
#endif
}

// ---------------- helpers -------------------------------------------------------

__global__ void alif_zero_state(int* __restrict__ p, int n) {
    for (int i = blockIdx.x * blockDim.x + threadIdx.x; i < n; i += gridDim.x * blockDim.x)
        p[i] = 0;
}

// f32 -> f16 weight conversion; row-major [N, K] f16 serves this code's WMMA
// B fragments in contiguous k order (k = 16*hi + s, as A; one 32B load per lane per k-tile).
__global__ void alif_pack_weights(const float* __restrict__ Wh, const float* __restrict__ Wout,
                                  _Float16* __restrict__ Wh16, _Float16* __restrict__ Wout16) {
    const int stride = gridDim.x * blockDim.x;
    const int tid = blockIdx.x * blockDim.x + threadIdx.x;
    for (int i = tid; i < H_DIM * IH_DIM; i += stride) Wh16[i]   = (_Float16)Wh[i];
    for (int i = tid; i < O_DIM * H_DIM;  i += stride) Wout16[i] = (_Float16)Wout[i];
}

// device-wide sense-reversing barrier (all NBLK blocks resident & participate)
__device__ __forceinline__ void gbar(volatile int* cnt, volatile int* gen) {
    __syncthreads();
    if (threadIdx.x == 0) {
        __threadfence();
        int g = *gen;
        if (atomicAdd((int*)cnt, 1) == NBLK - 1) {
            *cnt = 0;
            __threadfence();
            atomicAdd((int*)gen, 1);
        } else {
            while (*gen == g) { __builtin_amdgcn_s_sleep(2); }
        }
        __threadfence();
    }
    __syncthreads();
}

// A-fragment (16x32 f16) from a row-major f16 matrix: row r fixed, K base kb.
__device__ __forceinline__ v16h load_a_f16(const _Float16* __restrict__ row, int kb, int hi) {
    v16h a;
    const v8h lo = *(const v8h*)(row + kb + hi * 16);
    const v8h h8 = *(const v8h*)(row + kb + hi * 16 + 8);
#pragma unroll
    for (int q = 0; q < 8; ++q) { a[q] = lo[q]; a[8 + q] = h8[q]; }
    return a;
}

// A-fragment from an f32 row in LDS (vectorized ds_load_b128 + cvt)
__device__ __forceinline__ v16h load_a_f32_lds(const float* row, int kb, int hi) {
    const v4f a0 = *(const v4f*)(row + kb + hi * 16);
    const v4f a1 = *(const v4f*)(row + kb + hi * 16 + 4);
    const v4f a2 = *(const v4f*)(row + kb + hi * 16 + 8);
    const v4f a3 = *(const v4f*)(row + kb + hi * 16 + 12);
    v16h a;
#pragma unroll
    for (int q = 0; q < 4; ++q) {
        a[q]      = (_Float16)a0[q];
        a[4 + q]  = (_Float16)a1[q];
        a[8 + q]  = (_Float16)a2[q];
        a[12 + q] = (_Float16)a3[q];
    }
    return a;
}

// ---------------- persistent ALIF kernel ---------------------------------------
__global__ __launch_bounds__(TPB, 1)
void alif_persistent(const float* __restrict__ x,
                     const float* __restrict__ alpha_h,
                     const float* __restrict__ rho,
                     const float* __restrict__ beta,
                     const float* __restrict__ alpha_out,
                     const _Float16* __restrict__ Wh16,
                     const _Float16* __restrict__ Wout16,
                     _Float16* __restrict__ z0,
                     _Float16* __restrict__ z1,
                     float* __restrict__ u,
                     float* __restrict__ a_state,
                     float* __restrict__ ou,
                     float* __restrict__ out,
                     int* __restrict__ bar) {
    // double-buffered x tile: [2][16 rows][256 cols] f32 = 32 KB LDS
    __shared__ float lds_x[2][16 * I_DIM];

    const int lane  = threadIdx.x & 31;
    const int wave  = threadIdx.x >> 5;
    const int gwave = blockIdx.x * WPB + wave;
    const int c  = lane & 15;   // A row / B column / C column
    const int hi = lane >> 4;   // K sub-group select, C row-half select

    // phase A tile: 16 x 32 tiles over [B, H]; a_bt is uniform per block
    const int a_bt = gwave >> 5;
    const int a_ht = gwave & 31;
    const int blk_bt = blockIdx.x >> 2;          // == a_bt for all waves in block
    // phase B tile: 16 x 8 tiles over [B, O] (waves 0..127)
    const int b_bt = gwave >> 3;
    const int b_ot = gwave & 7;

    const int   hA = a_ht * 16 + c;
    const float al = alpha_h[hA];
    const float rh = rho[hA];
    const float be = beta[hA];
    const int   oB = b_ot * 16 + c;
    const float alo = alpha_out[oB];

    volatile int* bcnt = bar;
    volatile int* bgen = bar + 16;

    // issue async prefetch of x[tn] rows [blk_bt*16, +16) into LDS buffer `buf`
    auto issue_prefetch = [&](int tn, int buf) {
#pragma unroll
        for (int j = 0; j < 4; ++j) {
            const int chunk = threadIdx.x + j * TPB;          // 0..1023 b128 chunks
            const int row   = chunk >> 6;                     // 16 rows
            const int col   = (chunk & 63) * 4;               // 64 float4 per row
            const float* g = x + ((size_t)tn * B_DIM + blk_bt * 16 + row) * I_DIM + col;
            async_cp_b128(g, &lds_x[buf][row * I_DIM + col]);
        }
    };

    // prologue: stage x[0]
    issue_prefetch(0, 0);
    wait_async(0);
    __syncthreads();

    for (int t = 0; t < T_STEPS; ++t) {
        const _Float16* __restrict__ zprev = (t & 1) ? z1 : z0;
        _Float16* __restrict__       znext = (t & 1) ? z0 : z1;

        // kick off next step's x tile; overlaps all of step t's compute
        const int tn = (t + 1 < T_STEPS) ? t + 1 : t;
        issue_prefetch(tn, (t + 1) & 1);
        wait_async(4);           // batch t (older, in-order) complete; batch t+1 in flight
        __syncthreads();         // all waves' batch-t writes visible to the block

        // ---------------- phase A: cur = [x_t | z_prev] @ W_h^T, ALIF update ----
        {
            v8f acc = {};
            const int brow = a_bt * 16 + c;
            const float*    xrow = &lds_x[t & 1][c * I_DIM];    // this lane's A row
            const _Float16* zrow = zprev + (size_t)brow * H_DIM;
            const _Float16* wrow = Wh16 + (size_t)hA * IH_DIM;

            for (int k = 0; k < I_DIM; k += 32) {
                v16h af = load_a_f32_lds(xrow, k, hi);
                v16h bf = *(const v16h*)(wrow + k + hi * 16);
                acc = __builtin_amdgcn_wmma_f32_16x16x32_f16(false, af, false, bf,
                                                             (short)0, acc, false, false);
            }
            for (int k = 0; k < H_DIM; k += 32) {
                v16h af = load_a_f16(zrow, k, hi);
                v16h bf = *(const v16h*)(wrow + I_DIM + k + hi * 16);
                acc = __builtin_amdgcn_wmma_f32_16x16x32_f16(false, af, false, bf,
                                                             (short)0, acc, false, false);
            }

            // fused elementwise ALIF update directly on the C-fragment
#pragma unroll
            for (int i = 0; i < 8; ++i) {
                const int b = a_bt * 16 + hi * 8 + i;
                const size_t idx = (size_t)b * H_DIM + hA;
                const float zo = (float)zprev[idx];
                const float un = al * u[idx] + acc[i] - zo;               // soft reset
                const float an = rh * a_state[idx] + (1.0f - rh) * zo;    // adaptation
                const float thr = 1.0f + be * an;
                const float zn = (un - thr) > 0.0f ? 1.0f : 0.0f;         // spike
                u[idx] = un;
                a_state[idx] = an;
                znext[idx] = (_Float16)zn;
            }
        }

        // one device-wide barrier per step: z_next fully written before readout
        gbar(bcnt, bgen);

        // ---------------- phase B: ou = alpha_out*ou + z_next @ W_out^T ---------
        if (gwave < (B_DIM / 16) * (O_DIM / 16)) {
            v8f acc = {};
            const int brow = b_bt * 16 + c;
            const _Float16* zrow = znext + (size_t)brow * H_DIM;
            const _Float16* wrow = Wout16 + (size_t)oB * H_DIM;

            for (int k = 0; k < H_DIM; k += 32) {
                v16h af = load_a_f16(zrow, k, hi);
                v16h bf = *(const v16h*)(wrow + k + hi * 16);
                acc = __builtin_amdgcn_wmma_f32_16x16x32_f16(false, af, false, bf,
                                                             (short)0, acc, false, false);
            }
#pragma unroll
            for (int i = 0; i < 8; ++i) {
                const int b = b_bt * 16 + hi * 8 + i;
                const size_t idx = (size_t)b * O_DIM + oB;
                const float on = alo * ou[idx] + acc[i];
                ou[idx] = on;
                out[(size_t)t * B_DIM * O_DIM + idx] = on;
            }
        }
        // no second barrier needed: A(t+1) writes the other z buffer and u/a,
        // none of which phase B(t) touches.
    }
}

// ---------------- host launcher -------------------------------------------------
extern "C" void kernel_launch(void* const* d_in, const int* in_sizes, int n_in,
                              void* d_out, int out_size, void* d_ws, size_t ws_size,
                              hipStream_t stream) {
    const float* x         = (const float*)d_in[0];
    const float* W_h       = (const float*)d_in[1];
    const float* alpha_h   = (const float*)d_in[2];
    const float* rho       = (const float*)d_in[3];
    const float* beta      = (const float*)d_in[4];
    const float* W_out     = (const float*)d_in[5];
    const float* alpha_out = (const float*)d_in[6];
    float* out = (float*)d_out;

    char* ws = (char*)d_ws;
    int*      bar    = (int*)(ws + OFF_BAR);
    float*    u      = (float*)(ws + OFF_U);
    float*    a      = (float*)(ws + OFF_A);
    float*    ou     = (float*)(ws + OFF_OU);
    _Float16* z0     = (_Float16*)(ws + OFF_Z0);
    _Float16* z1     = (_Float16*)(ws + OFF_Z1);
    _Float16* Wh16   = (_Float16*)(ws + OFF_WH);
    _Float16* Wout16 = (_Float16*)(ws + OFF_WO);

    // zero barrier + all recurrent state (u, a, ou, z0, z1) every call
    const int zero_ints = (int)(OFF_WH / 4);
    alif_zero_state<<<512, 256, 0, stream>>>((int*)ws, zero_ints);

    // convert weights to f16 WMMA-friendly layout (row-major [N, K])
    alif_pack_weights<<<512, 256, 0, stream>>>(W_h, W_out, Wh16, Wout16);

    // single persistent kernel runs the whole T=1024 recurrence
    alif_persistent<<<NBLK, TPB, 0, stream>>>(x, alpha_h, rho, beta, alpha_out,
                                              Wh16, Wout16, z0, z1, u, a, ou, out, bar);
}
